// Informer_38259568673269
// MI455X (gfx1250) — matrix operand order repaired and checked by reading, not compiled, not run on hardware
//
#include <hip/hip_runtime.h>
#include <hip/hip_bf16.h>
#include <math.h>

// ---------------------------------------------------------------------------
// Informer (encoder/decoder transformer with ProbSparse top-k attention)
// CDNA5 / gfx1250.  Dense math via v_wmma_f32_16x16x32_f16 (f16 in, f32 acc).
// GEMM: block = 8 waves = 128(M) x 32(N) tile.  B tile staged into LDS in
// WMMA B-fragment order (1x global_load_b128 per thread per K-step, single
// ds_store_b64 when k-contiguous), double-buffered, one barrier per K-step;
// each wave reads its two B fragments with 2x ds_load_b128; A read directly
// with global_load_b128 (K contiguous).  Epilogues (bias, residual, ReLU,
// 1/sqrt(dk)) fused; all 32 (b,h) attention GEMMs batched via gridDim.z;
// cross-attn K/V hoisted out of the decoder loop.
// ---------------------------------------------------------------------------

typedef __attribute__((ext_vector_type(16))) _Float16 v16h;
typedef __attribute__((ext_vector_type(4)))  _Float16 v4h;
typedef __attribute__((ext_vector_type(8)))  float    v8f;

#define D_MODEL 512
#define N_HEADS 8
#define DK      64
#define D_FF    2048
#define BATCH   4
#define LSRC    1024
#define LTGT    512
#define IN_DIM  7
#define OUT_DIM 7

// ---------------------------------------------------------------------------
// Batched GEMM: C[z][m,n] = act( alpha * sum_k A[z][m,k]*B[z][k,n]
//                                + bias[n] + res[z][m,n] )
// A element (m,k) at A + zb*sAb + zh*sAh + m*lda + k        (k contiguous)
// B element (k,n) at B + zb*sBb + zh*sBh + k*ldbk + n*ldbn  (one of the two
//                                                            strides is 1)
// Requirements met by every call site: M % 128 == 0, N % 32 == 0, K % 32 == 0.
// ---------------------------------------------------------------------------
__global__ __launch_bounds__(256)
void gemm_wmma(const float* __restrict__ A, long long sAb, long long sAh, int lda,
               const float* __restrict__ B, long long sBb, long long sBh, int ldbk, int ldbn,
               float*       __restrict__ C, long long sCb, long long sCh, int ldc,
               const float* __restrict__ bias, const float* __restrict__ res,
               int M, int N, int Kd, int nh, float alpha, int relu)
{
    // Double-buffered 32(K) x 32(N) f16 B tile in WMMA B-fragment order:
    // halfIdx = ntileSel*512 + lane*16 + h, with k = h + 16*(lane>>4),
    // n = lane & 15  (contiguous k order, same as the A fragment).
    __shared__ _Float16 lds[2 * 1024];

    const int lane   = threadIdx.x & 31;
    const int wave   = threadIdx.x >> 5;
    const int laneHi = lane >> 4;
    const int lm     = lane & 15;
    const int mtile  = blockIdx.y * 128 + wave * 16;   // M % 128 == 0 always
    const int ntile  = blockIdx.x * 32;

    const int zb = blockIdx.z / nh;
    const int zh = blockIdx.z % nh;

    const float* Ap = A + zb * sAb + zh * sAh + (size_t)(mtile + lm) * lda;
    const float* Bt = B + zb * sBb + zh * sBh;

    // Per-thread B staging: 4 contiguous elements along the unit-stride dim
    // (one float4 load), addresses strength-reduced across the K loop.
    int k0, n0;
    if (ldbn == 1) { k0 = threadIdx.x >> 3; n0 = (threadIdx.x & 7) << 2; } // n-contig
    else           { n0 = threadIdx.x >> 3; k0 = (threadIdx.x & 7) << 2; } // k-contig
    const float* bptr  = Bt + (size_t)k0 * ldbk + (size_t)(ntile + n0) * ldbn;
    const size_t stepB = (size_t)32 * ldbk;
    const int    lbase = ((n0 >> 4) << 9) + (((n0 & 15) + (k0 & 16)) << 4) + (k0 & 15);

    v8f acc0 = {}, acc1 = {};
    int buf = 0;
    for (int kb = 0; kb < Kd; kb += 32) {
        // ---- stage B tile: one b128 load + fragment-order f16 stores ------
        float4 bv = *(const float4*)bptr;
        bptr += stepB;
        _Float16* L = &lds[buf << 10];
        if (ldbn == 1) {            // 4 consecutive n: same h, lanes L..L+3
            L[lbase]      = (_Float16)bv.x;
            L[lbase + 16] = (_Float16)bv.y;
            L[lbase + 32] = (_Float16)bv.z;
            L[lbase + 48] = (_Float16)bv.w;
        } else {                    // 4 consecutive k: consecutive h -> b64
            v4h p;
            p[0] = (_Float16)bv.x; p[1] = (_Float16)bv.y;
            p[2] = (_Float16)bv.z; p[3] = (_Float16)bv.w;
            *(v4h*)&L[lbase] = p;
        }
        __syncthreads();

        // ---- A 16x32 f16 fragment, direct from global (b128-vectorized) ---
        v16h a;
        #pragma unroll
        for (int j = 0; j < 8; ++j) {
            int kA = kb + 2 * j + (laneHi ? 16 : 0);
            a[2 * j]     = (_Float16)Ap[kA];
            a[2 * j + 1] = (_Float16)Ap[kA + 1];
        }
        if (kb + 32 < Kd) __builtin_prefetch(Ap + kb + 64, 0, 0); // global_prefetch_b8

        // ---- B fragments: 2x ds_load_b128 each ----------------------------
        v16h b0 = *(const v16h*)&lds[(buf << 10) + lane * 16];
        v16h b1 = *(const v16h*)&lds[(buf << 10) + 512 + lane * 16];

        acc0 = __builtin_amdgcn_wmma_f32_16x16x32_f16(false, a, false, b0,
                                                      (short)0, acc0, false, false);
        acc1 = __builtin_amdgcn_wmma_f32_16x16x32_f16(false, a, false, b1,
                                                      (short)0, acc1, false, false);
        buf ^= 1;
    }

    // C/D 16x16 f32 layout: lane n = lm, VGPR j -> m = j + 8*laneHi
    const long long co = zb * sCb + zh * sCh;
    #pragma unroll
    for (int j = 0; j < 8; ++j) {
        int m  = mtile + j + laneHi * 8;
        int n0c = ntile + lm;
        int n1c = ntile + 16 + lm;
        float v0 = acc0[j] * alpha;
        float v1 = acc1[j] * alpha;
        if (bias) { v0 += bias[n0c]; v1 += bias[n1c]; }
        if (res)  { v0 += res[co + (size_t)m * ldc + n0c];
                    v1 += res[co + (size_t)m * ldc + n1c]; }
        if (relu) { v0 = fmaxf(v0, 0.0f); v1 = fmaxf(v1, 0.0f); }
        C[co + (size_t)m * ldc + n0c] = v0;
        C[co + (size_t)m * ldc + n1c] = v1;
    }
}

// ---------------------------------------------------------------------------
// Embedding (K=7, scalar) + fused sinusoidal positional encoding.
// ---------------------------------------------------------------------------
__global__ void embed_pe(const float* __restrict__ in, const float* __restrict__ W,
                         const float* __restrict__ bias, float* __restrict__ out,
                         int L, int rows)
{
    int idx = blockIdx.x * blockDim.x + threadIdx.x;
    if (idx >= rows * D_MODEL) return;
    int m = idx / D_MODEL, d = idx % D_MODEL;
    int l = m % L;
    float acc = bias[d];
    #pragma unroll
    for (int i = 0; i < IN_DIM; ++i) acc += in[m * IN_DIM + i] * W[i * D_MODEL + d];
    int   p    = d >> 1;
    float freq = __expf(-(float)(2 * p) * (9.210340371976184f / (float)D_MODEL)); // ln(1e4)
    float ang  = (float)l * freq;
    acc += (d & 1) ? __cosf(ang) : __sinf(ang);
    out[idx] = acc;
}

// ---------------------------------------------------------------------------
// Row LayerNorm over D_MODEL: one wave32 per row, shuffle reductions.
// ---------------------------------------------------------------------------
__global__ __launch_bounds__(256)
void layernorm_k(const float* __restrict__ x, float* __restrict__ y,
                 const float* __restrict__ g, const float* __restrict__ b, int rows)
{
    int row = blockIdx.x * 8 + (threadIdx.x >> 5);
    if (row >= rows) return;
    int lane = threadIdx.x & 31;
    const float* xr = x + (size_t)row * D_MODEL;
    float v[16], s = 0.f, s2 = 0.f;
    #pragma unroll
    for (int i = 0; i < 16; ++i) { float t = xr[lane + 32 * i]; v[i] = t; s += t; s2 += t * t; }
    #pragma unroll
    for (int o = 16; o > 0; o >>= 1) { s += __shfl_xor(s, o, 32); s2 += __shfl_xor(s2, o, 32); }
    float mu  = s  * (1.0f / D_MODEL);
    float var = s2 * (1.0f / D_MODEL) - mu * mu;
    float r   = rsqrtf(var + 1e-5f);
    float* yr = y + (size_t)row * D_MODEL;
    #pragma unroll
    for (int i = 0; i < 16; ++i) { int d = lane + 32 * i; yr[d] = (v[i] - mu) * r * g[d] + b[d]; }
}

// ---------------------------------------------------------------------------
// ProbSparse top-k softmax, in place.  One wave32 per score row (Lk <= 1024,
// row lives in registers).  k-th largest found by 30-step value bisection
// with shuffle count-reductions; softmax over values >= threshold, rest 0
// (matches the reference scatter of softmaxed top-k probs into zeros).
// ---------------------------------------------------------------------------
__global__ __launch_bounds__(256)
void topk_softmax(float* __restrict__ S, int rows, int Lk, int topk)
{
    int row = blockIdx.x * 8 + (threadIdx.x >> 5);
    if (row >= rows) return;
    int lane = threadIdx.x & 31;
    float* sr = S + (size_t)row * Lk;
    const int nc = Lk >> 5;                   // 16 or 32
    float v[32];
    float mx = -3.4e38f, mn = 3.4e38f;
    for (int i = 0; i < nc; ++i) {
        float t = sr[lane + 32 * i];
        v[i] = t; mx = fmaxf(mx, t); mn = fminf(mn, t);
    }
    #pragma unroll
    for (int o = 16; o > 0; o >>= 1) {
        mx = fmaxf(mx, __shfl_xor(mx, o, 32));
        mn = fminf(mn, __shfl_xor(mn, o, 32));
    }
    float lo = mn, hi = mx;
    for (int it = 0; it < 30; ++it) {
        float mid = 0.5f * (lo + hi);
        int c = 0;
        for (int i = 0; i < nc; ++i) c += (v[i] >= mid) ? 1 : 0;
        #pragma unroll
        for (int o = 16; o > 0; o >>= 1) c += __shfl_xor(c, o, 32);
        if (c >= topk) lo = mid; else hi = mid;
    }
    float thr = lo, ssum = 0.f;
    for (int i = 0; i < nc; ++i) if (v[i] >= thr) ssum += __expf(v[i] - mx);
    #pragma unroll
    for (int o = 16; o > 0; o >>= 1) ssum += __shfl_xor(ssum, o, 32);
    float inv = 1.0f / ssum;
    for (int i = 0; i < nc; ++i) {
        int k = lane + 32 * i;
        sr[k] = (v[i] >= thr) ? __expf(v[i] - mx) * inv : 0.0f;
    }
}

// ---------------------------------------------------------------------------
// Output projection (N=7, scalar — not WMMA-shaped, negligible work).
// ---------------------------------------------------------------------------
__global__ void proj_out(const float* __restrict__ X, const float* __restrict__ W,
                         const float* __restrict__ bias, float* __restrict__ out, int rows)
{
    int idx = blockIdx.x * blockDim.x + threadIdx.x;
    if (idx >= rows * OUT_DIM) return;
    int m = idx / OUT_DIM, n = idx % OUT_DIM;
    float acc = bias[n];
    const float* xr = X + (size_t)m * D_MODEL;
    for (int k = 0; k < D_MODEL; ++k) acc += xr[k] * W[k * OUT_DIM + n];
    out[idx] = acc;
}

// ---------------------------------------------------------------------------
static inline void launch_gemm(hipStream_t st,
    const float* A, long long sAb, long long sAh, int lda,
    const float* B, long long sBb, long long sBh, int ldbk, int ldbn,
    float* C, long long sCb, long long sCh, int ldc,
    const float* bias, const float* res,
    int M, int N, int Kd, int nb, int nh, float alpha, int relu)
{
    dim3 grid(N / 32, M / 128, nb * nh);
    gemm_wmma<<<grid, dim3(256), 0, st>>>(A, sAb, sAh, lda, B, sBb, sBh, ldbk, ldbn,
                                          C, sCb, sCh, ldc, bias, res,
                                          M, N, Kd, nh, alpha, relu);
}

extern "C" void kernel_launch(void* const* d_in, const int* in_sizes, int n_in,
                              void* d_out, int out_size, void* d_ws, size_t ws_size,
                              hipStream_t stream)
{
    (void)in_sizes; (void)n_in; (void)out_size; (void)ws_size;
    // setup_inputs() dict insertion order, flattened depth-first (w then b;
    // attn dicts in q,k,v,o order; ln dicts g then b).
    const float* src  = (const float*)d_in[0];
    const float* tgt  = (const float*)d_in[1];
    const float* eW   = (const float*)d_in[2];  const float* eB   = (const float*)d_in[3];
    const float* dW   = (const float*)d_in[4];  const float* dB   = (const float*)d_in[5];
    const float* eqw  = (const float*)d_in[6];  const float* eqb  = (const float*)d_in[7];
    const float* ekw  = (const float*)d_in[8];  const float* ekb  = (const float*)d_in[9];
    const float* evw  = (const float*)d_in[10]; const float* evb  = (const float*)d_in[11];
    const float* eow  = (const float*)d_in[12]; const float* eob  = (const float*)d_in[13];
    const float* eln1g= (const float*)d_in[14]; const float* eln1b= (const float*)d_in[15];
    const float* ef1w = (const float*)d_in[16]; const float* ef1b = (const float*)d_in[17];
    const float* ef2w = (const float*)d_in[18]; const float* ef2b = (const float*)d_in[19];
    const float* eln2g= (const float*)d_in[20]; const float* eln2b= (const float*)d_in[21];
    const float* sqw  = (const float*)d_in[22]; const float* sqb  = (const float*)d_in[23];
    const float* skw  = (const float*)d_in[24]; const float* skb  = (const float*)d_in[25];
    const float* svw  = (const float*)d_in[26]; const float* svb  = (const float*)d_in[27];
    const float* sow  = (const float*)d_in[28]; const float* sob  = (const float*)d_in[29];
    const float* dln1g= (const float*)d_in[30]; const float* dln1b= (const float*)d_in[31];
    const float* cqw  = (const float*)d_in[32]; const float* cqb  = (const float*)d_in[33];
    const float* ckw  = (const float*)d_in[34]; const float* ckb  = (const float*)d_in[35];
    const float* cvw  = (const float*)d_in[36]; const float* cvb  = (const float*)d_in[37];
    const float* cow  = (const float*)d_in[38]; const float* cob  = (const float*)d_in[39];
    const float* dln2g= (const float*)d_in[40]; const float* dln2b= (const float*)d_in[41];
    const float* df1w = (const float*)d_in[42]; const float* df1b = (const float*)d_in[43];
    const float* df2w = (const float*)d_in[44]; const float* df2b = (const float*)d_in[45];
    const float* dln3g= (const float*)d_in[46]; const float* dln3b= (const float*)d_in[47];
    const float* pw   = (const float*)d_in[48]; const float* pb   = (const float*)d_in[49];

    // Workspace layout (floats).  SC (scores) region doubles as the FF hidden
    // buffer — never live at the same time.  Total ~210 MB.
    const size_t SZ_E = (size_t)BATCH * LSRC * D_MODEL;   // 2,097,152
    const size_t SZ_D = (size_t)BATCH * LTGT * D_MODEL;   // 1,048,576
    float* ws = (float*)d_ws;
    float* XE = ws;                 float* YE = XE + SZ_E;
    float* XD = YE + SZ_E;          float* YD = XD + SZ_D;
    float* Qb = YD + SZ_D;          float* Kb = Qb + SZ_E;
    float* Vb = Kb + SZ_E;          float* KC = Vb + SZ_E;
    float* VC = KC + SZ_E;          float* CT = VC + SZ_E;
    float* SC = CT + SZ_E;          // 33,554,432 floats (>= FF hidden needs)
    float* FF = SC;

    const int ME = BATCH * LSRC;    // 4096 rows
    const int MD = BATCH * LTGT;    // 2048 rows
    const float ISQ = 0.125f;       // 1/sqrt(64)

    // ---------------- Encoder ----------------
    embed_pe<<<(ME * D_MODEL + 255) / 256, 256, 0, stream>>>(src, eW, eB, XE, LSRC, ME);
    for (int l = 0; l < 3; ++l) {
        launch_gemm(stream, XE,0,0,D_MODEL, eqw,0,0,D_MODEL,1, Qb,0,0,D_MODEL, eqb,nullptr, ME,D_MODEL,D_MODEL,1,1,1.f,0);
        launch_gemm(stream, XE,0,0,D_MODEL, ekw,0,0,D_MODEL,1, Kb,0,0,D_MODEL, ekb,nullptr, ME,D_MODEL,D_MODEL,1,1,1.f,0);
        launch_gemm(stream, XE,0,0,D_MODEL, evw,0,0,D_MODEL,1, Vb,0,0,D_MODEL, evb,nullptr, ME,D_MODEL,D_MODEL,1,1,1.f,0);
        // scores[b,h] = Q K^T / 8 : A=Q slice (lda=512), B(k,n)=K[n*512+k]
        launch_gemm(stream,
            Qb, (long long)LSRC*D_MODEL, DK, D_MODEL,
            Kb, (long long)LSRC*D_MODEL, DK, 1, D_MODEL,
            SC, (long long)N_HEADS*LSRC*LSRC, (long long)LSRC*LSRC, LSRC,
            nullptr, nullptr, LSRC, LSRC, DK, BATCH, N_HEADS, ISQ, 0);
        { int rows = BATCH * N_HEADS * LSRC;
          topk_softmax<<<(rows + 7) / 8, 256, 0, stream>>>(SC, rows, LSRC, LSRC / 2); }
        // ctx[b,h] = P V : B(k,n)=V[k*512 + h*64 + n]
        launch_gemm(stream,
            SC, (long long)N_HEADS*LSRC*LSRC, (long long)LSRC*LSRC, LSRC,
            Vb, (long long)LSRC*D_MODEL, DK, D_MODEL, 1,
            CT, (long long)LSRC*D_MODEL, DK, D_MODEL,
            nullptr, nullptr, LSRC, DK, LSRC, BATCH, N_HEADS, 1.f, 0);
        launch_gemm(stream, CT,0,0,D_MODEL, eow,0,0,D_MODEL,1, YE,0,0,D_MODEL, eob, XE, ME,D_MODEL,D_MODEL,1,1,1.f,0);
        layernorm_k<<<ME / 8, 256, 0, stream>>>(YE, XE, eln1g, eln1b, ME);
        launch_gemm(stream, XE,0,0,D_MODEL, ef1w,0,0,D_FF,1, FF,0,0,D_FF, ef1b,nullptr, ME,D_FF,D_MODEL,1,1,1.f,1);
        launch_gemm(stream, FF,0,0,D_FF, ef2w,0,0,D_MODEL,1, YE,0,0,D_MODEL, ef2b, XE, ME,D_MODEL,D_FF,1,1,1.f,0);
        layernorm_k<<<ME / 8, 256, 0, stream>>>(YE, XE, eln2g, eln2b, ME);
    }

    // ---------------- Decoder ----------------
    embed_pe<<<(MD * D_MODEL + 255) / 256, 256, 0, stream>>>(tgt, dW, dB, XD, LTGT, MD);
    // cross-attn K/V hoisted: shared params + fixed enc_out => loop invariant
    launch_gemm(stream, XE,0,0,D_MODEL, ckw,0,0,D_MODEL,1, KC,0,0,D_MODEL, ckb,nullptr, ME,D_MODEL,D_MODEL,1,1,1.f,0);
    launch_gemm(stream, XE,0,0,D_MODEL, cvw,0,0,D_MODEL,1, VC,0,0,D_MODEL, cvb,nullptr, ME,D_MODEL,D_MODEL,1,1,1.f,0);

    for (int l = 0; l < 2; ++l) {
        // --- ProbSparse self attention (Lq=Lk=512) ---
        launch_gemm(stream, XD,0,0,D_MODEL, sqw,0,0,D_MODEL,1, Qb,0,0,D_MODEL, sqb,nullptr, MD,D_MODEL,D_MODEL,1,1,1.f,0);
        launch_gemm(stream, XD,0,0,D_MODEL, skw,0,0,D_MODEL,1, Kb,0,0,D_MODEL, skb,nullptr, MD,D_MODEL,D_MODEL,1,1,1.f,0);
        launch_gemm(stream, XD,0,0,D_MODEL, svw,0,0,D_MODEL,1, Vb,0,0,D_MODEL, svb,nullptr, MD,D_MODEL,D_MODEL,1,1,1.f,0);
        launch_gemm(stream,
            Qb, (long long)LTGT*D_MODEL, DK, D_MODEL,
            Kb, (long long)LTGT*D_MODEL, DK, 1, D_MODEL,
            SC, (long long)N_HEADS*LTGT*LTGT, (long long)LTGT*LTGT, LTGT,
            nullptr, nullptr, LTGT, LTGT, DK, BATCH, N_HEADS, ISQ, 0);
        { int rows = BATCH * N_HEADS * LTGT;
          topk_softmax<<<(rows + 7) / 8, 256, 0, stream>>>(SC, rows, LTGT, LTGT / 2); }
        launch_gemm(stream,
            SC, (long long)N_HEADS*LTGT*LTGT, (long long)LTGT*LTGT, LTGT,
            Vb, (long long)LTGT*D_MODEL, DK, D_MODEL, 1,
            CT, (long long)LTGT*D_MODEL, DK, D_MODEL,
            nullptr, nullptr, LTGT, DK, LTGT, BATCH, N_HEADS, 1.f, 0);
        launch_gemm(stream, CT,0,0,D_MODEL, sow,0,0,D_MODEL,1, YD,0,0,D_MODEL, sob, XD, MD,D_MODEL,D_MODEL,1,1,1.f,0);
        layernorm_k<<<MD / 8, 256, 0, stream>>>(YD, XD, dln1g, dln1b, MD);

        // --- cross attention (Lq=512, Lk=1024) ---
        launch_gemm(stream, XD,0,0,D_MODEL, cqw,0,0,D_MODEL,1, Qb,0,0,D_MODEL, cqb,nullptr, MD,D_MODEL,D_MODEL,1,1,1.f,0);
        launch_gemm(stream,
            Qb, (long long)LTGT*D_MODEL, DK, D_MODEL,
            KC, (long long)LSRC*D_MODEL, DK, 1, D_MODEL,
            SC, (long long)N_HEADS*LTGT*LSRC, (long long)LTGT*LSRC, LSRC,
            nullptr, nullptr, LTGT, LSRC, DK, BATCH, N_HEADS, ISQ, 0);
        { int rows = BATCH * N_HEADS * LTGT;
          topk_softmax<<<(rows + 7) / 8, 256, 0, stream>>>(SC, rows, LSRC, LSRC / 2); }
        launch_gemm(stream,
            SC, (long long)N_HEADS*LTGT*LSRC, (long long)LTGT*LSRC, LSRC,
            VC, (long long)LSRC*D_MODEL, DK, D_MODEL, 1,
            CT, (long long)LTGT*D_MODEL, DK, D_MODEL,
            nullptr, nullptr, LTGT, DK, LSRC, BATCH, N_HEADS, 1.f, 0);
        launch_gemm(stream, CT,0,0,D_MODEL, cow,0,0,D_MODEL,1, YD,0,0,D_MODEL, cob, XD, MD,D_MODEL,D_MODEL,1,1,1.f,0);
        layernorm_k<<<MD / 8, 256, 0, stream>>>(YD, XD, dln2g, dln2b, MD);

        // --- feed-forward ---
        launch_gemm(stream, XD,0,0,D_MODEL, df1w,0,0,D_FF,1, FF,0,0,D_FF, df1b,nullptr, MD,D_FF,D_MODEL,1,1,1.f,1);
        launch_gemm(stream, FF,0,0,D_FF, df2w,0,0,D_MODEL,1, YD,0,0,D_MODEL, df2b, XD, MD,D_MODEL,D_FF,1,1,1.f,0);
        layernorm_k<<<MD / 8, 256, 0, stream>>>(YD, XD, dln3g, dln3b, MD);
    }

    proj_out<<<(MD * OUT_DIM + 255) / 256, 256, 0, stream>>>(XD, pw, pb, (float*)d_out, MD);
}
